// SpectralOT_Loss_20658792694558
// MI455X (gfx1250) — compile-verified, matrix operand order checked by reading, not run on hardware
//
#include <hip/hip_runtime.h>
#include <hip/hip_bf16.h>
#include <math.h>

// ---------------- problem constants ----------------
#define N_FFT      2048
#define HOP        512
#define BATCH      32
#define T_SAMPLES  176400
#define N_FRAMES   345                  // (T + 2*pad - N_FFT)/HOP + 1
#define F_BINS     1025                 // N_FFT/2 + 1
#define F_PAD      1040                 // 65 tiles * 16
#define ROWS_SIG   (BATCH * N_FRAMES)   // 11040 frames per signal (x block, then y block)
#define M_ROWS     (2 * ROWS_SIG)       // 22080 total GEMM rows
#define KDIM       N_FFT
#define W_ROWS     (2 * F_PAD)          // 1040 re rows + 1040 im rows

#define M_TILES    (M_ROWS / 16)        // 1380
#define MQ_TILES   (M_TILES / 4)        // 345 (each wave owns 4 M-tiles = 64 rows)
#define F_TILES    (F_PAD / 16)         // 65
#define NQ_TILES   (MQ_TILES * F_TILES) // 22425 wave-tiles

typedef __attribute__((ext_vector_type(16))) _Float16 v16h;
typedef __attribute__((ext_vector_type(8)))  _Float16 v8h;
typedef __attribute__((ext_vector_type(8)))  float    v8f;

union U16h { v16h v; v8h h[2]; };

// ---------------- phase 0: DFT matrix (freq-major, f16) ----------------
// WT[r][n], r in [0,1040): cos(2*pi*r*n/2048); r in [1040,2080): -sin(...) for f=r-1040.
// Rows with f > 1024 are zero padding.
__global__ void gen_w_kernel(_Float16* __restrict__ WT) {
    int idx = blockIdx.x * blockDim.x + threadIdx.x;
    if (idx >= W_ROWS * KDIM) return;
    int n = idx & (KDIM - 1);
    int r = idx >> 11;
    int isIm = (r >= F_PAD);
    int f = isIm ? (r - F_PAD) : r;
    float val = 0.0f;
    if (f < F_BINS) {
        int ph = (f * n) & (KDIM - 1);            // cos/sin period 2048 in integer phase
        float th = (float)ph * (6.283185307179586f / (float)N_FFT);
        float s, c;
        __sincosf(th, &s, &c);
        val = isIm ? -s : c;
    }
    WT[idx] = (_Float16)val;
}

// ---------------- phase 1: windowed frames (reflect pad), f16 ----------------
__global__ void build_a_kernel(_Float16* __restrict__ A,
                               const float* __restrict__ xsig,
                               const float* __restrict__ ysig) {
    int idx = blockIdx.x * blockDim.x + threadIdx.x;
    if (idx >= M_ROWS * KDIM) return;
    int n   = idx & (KDIM - 1);
    int row = idx >> 11;
    int sig = (row >= ROWS_SIG);                  // 0 = x block, 1 = y block
    int rr  = row - sig * ROWS_SIG;
    int b   = rr / N_FRAMES;
    int t   = rr - b * N_FRAMES;
    int orig = t * HOP + n - (N_FFT / 2);         // position in unpadded signal
    if (orig < 0) orig = -orig;                   // reflect (pad < T, single bounce)
    if (orig >= T_SAMPLES) orig = 2 * (T_SAMPLES - 1) - orig;
    const float* src = sig ? ysig : xsig;
    float w = 0.5f - 0.5f * __cosf((float)n * (6.283185307179586f / (float)N_FFT));
    A[idx] = (_Float16)(src[(size_t)b * T_SAMPLES + orig] * w);
}

// ---------------- phase 2: DFT GEMM via WMMA, fused |.| epilogue ----------------
// Register-blocked: one wave owns a 64x16 output slab (4 M-tiles) for BOTH the
// real and imaginary DFT halves: 8 WMMAs per K-step against 12 b128 loads.
// Epilogue fuses mag = sqrt(re^2 + im^2) -> MAG[M_ROWS][F_PAD].
__global__ void dft_wmma_kernel(const _Float16* __restrict__ A,
                                const _Float16* __restrict__ WT,
                                float* __restrict__ MAG) {
    int wid  = threadIdx.x >> 5;
    int lane = threadIdx.x & 31;
    int g = blockIdx.x * 8 + wid;                 // 8 waves / block
    if (g >= NQ_TILES) return;
    int tq = g / F_TILES;                         // M-quad tile (64 rows)
    int tf = g - tq * F_TILES;                    // freq tile

    int m    = lane & 15;                         // matrix row within 16-row tile
    int koff = (lane < 16) ? 0 : 8;               // ISA 16-bit A/B fragment layout

    const _Float16* arow[4];
    #pragma unroll
    for (int s = 0; s < 4; ++s)
        arow[s] = A + (size_t)(tq * 64 + s * 16 + m) * KDIM;
    const _Float16* brre = WT + (size_t)(tf * 16 + m) * KDIM;           // cos rows
    const _Float16* brim = WT + (size_t)(F_PAD + tf * 16 + m) * KDIM;   // -sin rows

    v8f cre[4], cim[4];
    #pragma unroll
    for (int s = 0; s < 4; ++s) { cre[s] = (v8f){}; cim[s] = (v8f){}; }

    for (int k = 0; k < KDIM; k += 32) {
        U16h bre, bim;
        bre.h[0] = *reinterpret_cast<const v8h*>(brre + k + koff);
        bre.h[1] = *reinterpret_cast<const v8h*>(brre + k + 16 + koff);
        bim.h[0] = *reinterpret_cast<const v8h*>(brim + k + koff);
        bim.h[1] = *reinterpret_cast<const v8h*>(brim + k + 16 + koff);
        U16h a[4];
        #pragma unroll
        for (int s = 0; s < 4; ++s) {
            a[s].h[0] = *reinterpret_cast<const v8h*>(arow[s] + k + koff);
            a[s].h[1] = *reinterpret_cast<const v8h*>(arow[s] + k + 16 + koff);
        }
        #pragma unroll
        for (int s = 0; s < 4; ++s) {
            cre[s] = __builtin_amdgcn_wmma_f32_16x16x32_f16(false, a[s].v, false, bre.v,
                                                            (short)0, cre[s], false, false);
            cim[s] = __builtin_amdgcn_wmma_f32_16x16x32_f16(false, a[s].v, false, bim.v,
                                                            (short)0, cim[s], false, false);
        }
    }

    // C/D layout: lane holds N = lane&15; VGPR v holds M = v + (lane<16 ? 0 : 8)
    int nc    = lane & 15;
    int mbase = (lane < 16) ? 0 : 8;
    #pragma unroll
    for (int s = 0; s < 4; ++s) {
        #pragma unroll
        for (int v = 0; v < 8; ++v) {
            float re = cre[s][v], im = cim[s][v];
            MAG[(size_t)(tq * 64 + s * 16 + mbase + v) * F_PAD + (size_t)(tf * 16 + nc)] =
                sqrtf(re * re + im * im);
        }
    }
}

// ---------------- phase 3: per-frame OT ----------------
__device__ inline int lbound1025(const float* a, float v) {
    int lo = 0, hi = F_BINS;
    while (lo < hi) { int mid = (lo + hi) >> 1; if (a[mid] < v) lo = mid + 1; else hi = mid; }
    return lo;                                     // first idx with a[idx] >= v ('left')
}

__global__ void zero_out_kernel(float* __restrict__ out) {
    if (threadIdx.x < BATCH) out[threadIdx.x] = 0.0f;
}

// 2 waves / block; one wave owns one (batch, frame) pair.
__global__ void ot_kernel(const float* __restrict__ MAG, float* __restrict__ out) {
    __shared__ float lds[2][2 * F_BINS + 2 * F_BINS];   // ax | ay | qs  (4100 floats/wave)
    int wid  = threadIdx.x >> 5;
    int lane = threadIdx.x & 31;
    int pair = blockIdx.x * 2 + wid;               // 5520*2 == 11040 exactly
    int b = pair / N_FRAMES;
    int t = pair - b * N_FRAMES;

    const float* rx = MAG + (size_t)(b * N_FRAMES + t) * F_PAD;
    const float* ry = MAG + (size_t)(ROWS_SIG + b * N_FRAMES + t) * F_PAD;
    float* ax = &lds[wid][0];
    float* ay = ax + F_BINS;
    float* qs = ay + F_BINS;                       // 2*F_BINS = 2050 merged quantiles

    // ---- normalized cumsum (CDF) of each magnitude row, shuffle scan ----
    #pragma unroll 1
    for (int pass = 0; pass < 2; ++pass) {
        const float* src = pass ? ry : rx;
        float* dst = pass ? ay : ax;
        float run = 0.0f;
        for (int base = 0; base < F_BINS; base += 32) {
            int i = base + lane;
            float v = (i < F_BINS) ? src[i] : 0.0f;
            #pragma unroll
            for (int d = 1; d < 32; d <<= 1) {
                float u = __shfl_up(v, d, 32);
                if (lane >= d) v += u;
            }
            float tot = __shfl(v, 31, 32);
            if (i < F_BINS) dst[i] = run + v;
            run += tot;
        }
        float inv = 1.0f / run;                    // run == dst[F_BINS-1] (normalize by last)
        for (int i = lane; i < F_BINS; i += 32) dst[i] *= inv;
    }
    __syncthreads();

    // ---- merge the two sorted CDFs (== jnp.sort of concat) ----
    if (lane == 0) {
        int i = 0, j = 0;
        for (int k = 0; k < 2 * F_BINS; ++k) {
            float vx = (i < F_BINS) ? ax[i] : 3.4e38f;
            float vy = (j < F_BINS) ? ay[j] : 3.4e38f;
            if (vx <= vy) { qs[k] = vx; ++i; } else { qs[k] = vy; ++j; }
        }
    }
    __syncthreads();

    // ---- parallel searchsorted + weighted sum over 2049 intervals ----
    float acc = 0.0f;
    for (int k = lane; k < 2 * F_BINS - 1; k += 32) {
        float q1 = qs[k + 1];
        float dr = q1 - qs[k];
        int ix = lbound1025(ax, q1); if (ix > F_BINS - 1) ix = F_BINS - 1;
        int iy = lbound1025(ay, q1); if (iy > F_BINS - 1) iy = F_BINS - 1;
        float d = (float)(iy - ix) * (1.0f / (float)N_FFT);   // rfftfreq: f[i] = i/N
        acc += d * d * dr;
    }
    #pragma unroll
    for (int d = 16; d >= 1; d >>= 1) acc += __shfl_xor(acc, d, 32);
    if (lane == 0) atomicAdd(&out[b], acc * (100.0f / (float)N_FRAMES));
}

// ---------------- launch ----------------
extern "C" void kernel_launch(void* const* d_in, const int* in_sizes, int n_in,
                              void* d_out, int out_size, void* d_ws, size_t ws_size,
                              hipStream_t stream) {
    const float* ysig = (const float*)d_in[0];     // setup_inputs order: 'y', 'x'
    const float* xsig = (const float*)d_in[1];
    float* out = (float*)d_out;
    char* ws = (char*)d_ws;

    const size_t wt_bytes  = (size_t)W_ROWS * KDIM * sizeof(_Float16);   // 8.5 MB
    const size_t a_bytes   = (size_t)M_ROWS * KDIM * sizeof(_Float16);   // 90.4 MB
    _Float16* WT  = (_Float16*)ws;
    _Float16* A   = (_Float16*)(ws + wt_bytes);
    float*    MAG = (float*)(ws + wt_bytes + a_bytes);                   // 91.9 MB

    {   // phase 0: DFT matrix
        int n = W_ROWS * KDIM;
        gen_w_kernel<<<(n + 255) / 256, 256, 0, stream>>>(WT);
    }
    {   // phase 1: windowed frames
        int n = M_ROWS * KDIM;
        build_a_kernel<<<(n + 255) / 256, 256, 0, stream>>>(A, xsig, ysig);
    }
    {   // phase 2: WMMA DFT + magnitude (register-blocked 64x16 x {re,im} per wave)
        int blocks = (NQ_TILES + 7) / 8;           // 8 waves per 256-thread block
        dft_wmma_kernel<<<blocks, 256, 0, stream>>>(A, WT, MAG);
    }
    zero_out_kernel<<<1, 32, 0, stream>>>(out);
    {   // phase 3: per-frame OT, atomic accumulate per batch
        ot_kernel<<<ROWS_SIG / 2, 64, 0, stream>>>(MAG, out);
    }
}
